// RegressionRNN_47442208752164
// MI455X (gfx1250) — compile-verified, matrix operand order checked by reading, not run on hardware
//
#include <hip/hip_runtime.h>
#include <hip/hip_bf16.h>

typedef __attribute__((ext_vector_type(16))) __bf16 v16bf;
typedef __attribute__((ext_vector_type(8)))  float  v8f;

#define B_ROWS   1024
#define T_STEPS  512
#define CELL     512
#define FEAT     24
#define KP_LSTM  544      // 24 x + 512 h + 8 pad, multiple of 32
#define DENSE    2048
#define PRED     576
#define PRED_P   640      // padded to multiple of 128

__device__ __forceinline__ __bf16 f2bf(float f) {
    union { float f; unsigned u; } x; x.f = f;
    unsigned r = (x.u + 0x7FFFu + ((x.u >> 16) & 1u)) >> 16;
    unsigned short s = (unsigned short)r;
    return __builtin_bit_cast(__bf16, s);
}
__device__ __forceinline__ float sigmf(float x) { return 1.0f / (1.0f + __expf(-x)); }
__device__ __forceinline__ float tanhf_fast(float x) { return 2.0f * sigmf(2.0f * x) - 1.0f; }

union FragU { v16bf v; uint4 q[2]; };

// ---------------------------------------------------------------------------
// LSTM step: z = [x_t | h] @ W + b (bf16 WMMA, fp32 acc) fused with gate math.
// Grid: (8 M-blocks, 16 cell-chunks). 256 threads = 8 waves; wave w owns rows
// [blk*128 + w*16, +16) and all 128 block-columns = 4 gates x 32 cells.
// ---------------------------------------------------------------------------
__global__ __launch_bounds__(256) void lstm_step_kernel(
    const __bf16* __restrict__ Acur,    // [1024][544] bf16  (x_t | h_t | 0)
    const __bf16* __restrict__ WT,      // [2048][544] bf16  (W_lstm transposed)
    const float*  __restrict__ b_lstm,  // [2048]
    float*        __restrict__ c_buf,   // [1024][512]
    __bf16*       __restrict__ Anext)   // [1024][544]  (writes h cols 24..535)
{
    __shared__ uint4 ldsB[8 * 32 * 2];          // 8 N-tiles, frag-ordered, 8KB
    const int tid  = threadIdx.x;
    const int lane = tid & 31;
    const int w    = tid >> 5;
    const int r16  = lane & 15;
    const int hi   = lane >> 4;
    const int cc   = blockIdx.y;                 // cell chunk (32 cells)
    const int rowbase = blockIdx.x * 128 + w * 16;
    const int AROW = rowbase + r16;
    const int akoff = hi * 8;

    v8f acc[8] = {};

    for (int ks = 0; ks < KP_LSTM / 32; ++ks) {
        const int k0 = ks * 32;
        // cooperative fill of B tile into LDS in fragment order
        #pragma unroll
        for (int i = 0; i < 2; ++i) {
            int cid  = tid * 2 + i;
            int half = cid & 1;
            int bl   = (cid >> 1) & 31;
            int nt   = cid >> 6;
            int rowB = (nt >> 1) * CELL + cc * 32 + (nt & 1) * 16 + (bl & 15);
            int koff = k0 + (bl >> 4) * 8 + half * 16;
            ldsB[cid] = *(const uint4*)(WT + (size_t)rowB * KP_LSTM + koff);
        }
        __syncthreads();
        // A fragment (16x32 bf16, per-lane layout per ISA 7.12.2)
        FragU fa;
        fa.q[0] = *(const uint4*)(Acur + (size_t)AROW * KP_LSTM + k0 + akoff);
        fa.q[1] = *(const uint4*)(Acur + (size_t)AROW * KP_LSTM + k0 + akoff + 16);
        if (ks + 1 < KP_LSTM / 32)  // WGP-scope prefetch of next A chunk
            __builtin_prefetch(Acur + (size_t)AROW * KP_LSTM + k0 + 32, 0, 3);
        #pragma unroll
        for (int nt = 0; nt < 8; ++nt) {
            FragU fb;
            fb.q[0] = ldsB[(nt * 32 + lane) * 2 + 0];
            fb.q[1] = ldsB[(nt * 32 + lane) * 2 + 1];
            acc[nt] = __builtin_amdgcn_wmma_f32_16x16x32_bf16(
                false, fa.v, false, fb.v, (short)0, acc[nt], false, false);
        }
        __syncthreads();
    }

    // Fused gate epilogue: tiles nt = g*2 + s hold gate g, cells s*16 + r16.
    #pragma unroll
    for (int s = 0; s < 2; ++s) {
        const int cell = cc * 32 + s * 16 + r16;
        const float bi = b_lstm[0 * CELL + cell];
        const float bj = b_lstm[1 * CELL + cell];
        const float bf = b_lstm[2 * CELL + cell];
        const float bo = b_lstm[3 * CELL + cell];
        #pragma unroll
        for (int j = 0; j < 8; ++j) {
            const int row = rowbase + j + 8 * hi;
            float iv = acc[0 + s][j] + bi;
            float jv = acc[2 + s][j] + bj;
            float fv = acc[4 + s][j] + bf + 1.0f;   // forget bias
            float ov = acc[6 + s][j] + bo;
            float cold = c_buf[(size_t)row * CELL + cell];
            float cnew = cold * sigmf(fv) + sigmf(iv) * tanhf_fast(jv);
            float hnew = tanhf_fast(cnew) * sigmf(ov);
            c_buf[(size_t)row * CELL + cell] = cnew;
            Anext[(size_t)row * KP_LSTM + FEAT + cell] = f2bf(hnew);
        }
    }
}

// ---------------------------------------------------------------------------
// Generic bf16 GEMM + bias (+optional relu), fp32 or bf16 output.
// C[1024, N] = A[1024, K] @ WT[N, K]^T. Grid: (8, N_P/128), 256 threads.
// ---------------------------------------------------------------------------
__global__ __launch_bounds__(256) void gemm_bias_kernel(
    const __bf16* __restrict__ A,  int lda,
    const __bf16* __restrict__ WT, int ldw,
    const float*  __restrict__ bias,
    float* __restrict__ outf, __bf16* __restrict__ outbf, int ldo,
    int Nreal, int ksteps, int relu)
{
    __shared__ uint4 ldsB[8 * 32 * 2];
    const int tid  = threadIdx.x;
    const int lane = tid & 31;
    const int w    = tid >> 5;
    const int r16  = lane & 15;
    const int hi   = lane >> 4;
    const int rowbase = blockIdx.x * 128 + w * 16;
    const int nbase   = blockIdx.y * 128;
    const int AROW = rowbase + r16;
    const int akoff = hi * 8;

    v8f acc[8] = {};

    for (int ks = 0; ks < ksteps; ++ks) {
        const int k0 = ks * 32;
        #pragma unroll
        for (int i = 0; i < 2; ++i) {
            int cid  = tid * 2 + i;
            int half = cid & 1;
            int bl   = (cid >> 1) & 31;
            int nt   = cid >> 6;
            int rowB = nbase + nt * 16 + (bl & 15);
            int koff = k0 + (bl >> 4) * 8 + half * 16;
            ldsB[cid] = *(const uint4*)(WT + (size_t)rowB * ldw + koff);
        }
        __syncthreads();
        FragU fa;
        fa.q[0] = *(const uint4*)(A + (size_t)AROW * lda + k0 + akoff);
        fa.q[1] = *(const uint4*)(A + (size_t)AROW * lda + k0 + akoff + 16);
        #pragma unroll
        for (int nt = 0; nt < 8; ++nt) {
            FragU fb;
            fb.q[0] = ldsB[(nt * 32 + lane) * 2 + 0];
            fb.q[1] = ldsB[(nt * 32 + lane) * 2 + 1];
            acc[nt] = __builtin_amdgcn_wmma_f32_16x16x32_bf16(
                false, fa.v, false, fb.v, (short)0, acc[nt], false, false);
        }
        __syncthreads();
    }

    #pragma unroll
    for (int nt = 0; nt < 8; ++nt) {
        const int col = nbase + nt * 16 + r16;
        if (col < Nreal) {
            const float bv = bias[col];
            #pragma unroll
            for (int j = 0; j < 8; ++j) {
                const int row = rowbase + j + 8 * hi;
                float v = acc[nt][j] + bv;
                if (relu) v = fmaxf(v, 0.0f);
                if (outbf) outbf[(size_t)row * ldo + col] = f2bf(v);
                else       outf [(size_t)row * ldo + col] = v;
            }
        }
    }
}

// --------------------------- setup / utility kernels -----------------------
__global__ void transpose_pad_kernel(const float* __restrict__ W, int K, int N,
                                     __bf16* __restrict__ WT, int KP, int NP) {
    int id = blockIdx.x * 256 + threadIdx.x;
    if (id >= NP * KP) return;
    int n = id / KP, k = id % KP;
    float v = (n < N && k < K) ? W[(size_t)k * N + n] : 0.0f;
    WT[id] = f2bf(v);
}

__global__ void pack_x_kernel(const float* __restrict__ X, int t,
                              __bf16* __restrict__ Adst) {
    int id = blockIdx.x * 256 + threadIdx.x;
    if (id >= B_ROWS * FEAT) return;
    int row = id / FEAT, col = id % FEAT;
    Adst[(size_t)row * KP_LSTM + col] =
        f2bf(X[((size_t)row * T_STEPS + t) * FEAT + col]);
}

__global__ void zero32_kernel(unsigned* __restrict__ p, int n) {
    int id = blockIdx.x * 256 + threadIdx.x;
    if (id < n) p[id] = 0u;
}

// ---------------------------------------------------------------------------
extern "C" void kernel_launch(void* const* d_in, const int* in_sizes, int n_in,
                              void* d_out, int out_size, void* d_ws, size_t ws_size,
                              hipStream_t stream) {
    const float* X       = (const float*)d_in[0];
    const float* W_lstm  = (const float*)d_in[1];
    const float* b_lstm  = (const float*)d_in[2];
    const float* W_dense = (const float*)d_in[3];
    const float* b_dense = (const float*)d_in[4];
    const float* W_pred  = (const float*)d_in[5];
    const float* b_pred  = (const float*)d_in[6];
    float* out = (float*)d_out;

    char* ws = (char*)d_ws;
    __bf16* WT_lstm  = (__bf16*)ws; ws += (size_t)4 * CELL * KP_LSTM * 2;  // 2.23 MB
    __bf16* WT_dense = (__bf16*)ws; ws += (size_t)DENSE * CELL * 2;        // 2.10 MB
    __bf16* WT_pred  = (__bf16*)ws; ws += (size_t)PRED_P * DENSE * 2;      // 2.62 MB
    __bf16* A0       = (__bf16*)ws; ws += (size_t)B_ROWS * KP_LSTM * 2;    // 1.11 MB
    __bf16* A1       = (__bf16*)ws; ws += (size_t)B_ROWS * KP_LSTM * 2;    // 1.11 MB
    float*  c_buf    = (float*)ws;  ws += (size_t)B_ROWS * CELL * 4;       // 2.10 MB
    __bf16* dense_b  = (__bf16*)ws;                                        // 4.19 MB

    // One-time prep: bf16 transposed/padded weights, zeroed state.
    {
        int n;
        n = 4 * CELL * KP_LSTM;
        transpose_pad_kernel<<<(n + 255) / 256, 256, 0, stream>>>(
            W_lstm, FEAT + CELL, 4 * CELL, WT_lstm, KP_LSTM, 4 * CELL);
        n = DENSE * CELL;
        transpose_pad_kernel<<<(n + 255) / 256, 256, 0, stream>>>(
            W_dense, CELL, DENSE, WT_dense, CELL, DENSE);
        n = PRED_P * DENSE;
        transpose_pad_kernel<<<(n + 255) / 256, 256, 0, stream>>>(
            W_pred, DENSE, PRED, WT_pred, DENSE, PRED_P);
        n = B_ROWS * CELL;
        zero32_kernel<<<(n + 255) / 256, 256, 0, stream>>>((unsigned*)c_buf, n);
        n = B_ROWS * KP_LSTM / 2;  // bf16 pairs as u32
        zero32_kernel<<<(n + 255) / 256, 256, 0, stream>>>((unsigned*)A0, n);
        zero32_kernel<<<(n + 255) / 256, 256, 0, stream>>>((unsigned*)A1, n);
    }

    __bf16* Ab[2] = { A0, A1 };
    const int npack = B_ROWS * FEAT;
    pack_x_kernel<<<(npack + 255) / 256, 256, 0, stream>>>(X, 0, A0);

    for (int t = 0; t < T_STEPS; ++t) {
        if (t > 0)
            pack_x_kernel<<<(npack + 255) / 256, 256, 0, stream>>>(X, t, Ab[t & 1]);
        lstm_step_kernel<<<dim3(B_ROWS / 128, CELL / 32), 256, 0, stream>>>(
            Ab[t & 1], WT_lstm, b_lstm, c_buf, Ab[(t + 1) & 1]);
    }
    // final h is in Ab[T&1] = A0, columns [24, 536)

    // dense = relu(h @ W_dense + b_dense), bf16 out
    gemm_bias_kernel<<<dim3(B_ROWS / 128, DENSE / 128), 256, 0, stream>>>(
        A0 + FEAT, KP_LSTM, WT_dense, CELL, b_dense,
        nullptr, dense_b, DENSE, DENSE, CELL / 32, 1);

    // preds = dense @ W_pred + b_pred, fp32 out
    gemm_bias_kernel<<<dim3(B_ROWS / 128, PRED_P / 128), 256, 0, stream>>>(
        dense_b, DENSE, WT_pred, DENSE, b_pred,
        out, nullptr, PRED, PRED, DENSE / 32, 0);
}
